// RVQVAE_28415503630583
// MI455X (gfx1250) — matrix operand order repaired and checked by reading, not compiled, not run on hardware
//
#include <hip/hip_runtime.h>
#include <hip/hip_bf16.h>
#include <math.h>

// ---------------------------------------------------------------------------
// RVQ-VAE forward for gfx1250 (MI455X). All GEMM-shaped work (conv1d layers,
// VQ distance computation) runs through V_WMMA_F32_16X16X32_F16 (f16 in, f32
// accumulate). Activations are channels-last f16; weights are pre-packed to
// (C_out, K, C_in_pad) f16 so every WMMA fragment load is a contiguous v8h.
// Conv waves are register-tiled 64(M)x16(N): 4 accumulators share one im2col
// B fragment per K-step (amortizes the gather 4x, 4 independent WMMA chains).
// ---------------------------------------------------------------------------

typedef __attribute__((ext_vector_type(16))) _Float16 v16h;
typedef __attribute__((ext_vector_type(8)))  _Float16 v8h;
typedef __attribute__((ext_vector_type(8)))  float    v8f;

#define NBATCH   64
#define TFULL    256
#define INW      263
#define INWP     264
#define WIDTH    512
#define CODE_DIM 512
#define NB_CODE  1024
#define NUM_Q    6
#define TQ       32
#define NTOK     (NBATCH * TQ)          // 2048 tokens
#define XOUT_N   (NBATCH * TFULL * INW) // 4,308,992 floats of x_out

// ------------------------------- prep kernels ------------------------------

// (C_out, C_in, Kw) f32  ->  (C_out, Kw, C_in_pad) f16, zero-padded channels
__global__ void prep_weight(const float* __restrict__ w, _Float16* __restrict__ wp,
                            int Cout, int Cin, int Kw, int Cinp, long total) {
    long i = (long)blockIdx.x * 256 + threadIdx.x;
    if (i >= total) return;
    int  ci = (int)(i % Cinp);
    long r  = i / Cinp;
    int  kw = (int)(r % Kw);
    int  co = (int)(r / Kw);
    float v = (ci < Cin) ? w[((long)co * Cin + ci) * Kw + kw] : 0.f;
    wp[i] = (_Float16)v;
}

// x (B,T,263) f32 -> packed (B,T,264) f16 with zero pad channel
__global__ void pack_x(const float* __restrict__ x, _Float16* __restrict__ xp, long total) {
    long i = (long)blockIdx.x * 256 + threadIdx.x;
    if (i >= total) return;
    int  c  = (int)(i % INWP);
    long nt = i / INWP;
    xp[i] = (c < INW) ? (_Float16)x[nt * INW + c] : (_Float16)0.f;
}

// codebooks f32 -> f16 copy + per-code squared norm. blockIdx over (q*1024+code)
__global__ void prep_cb(const float* __restrict__ cb, _Float16* __restrict__ cbh,
                        float* __restrict__ cnorm) {
    int b = blockIdx.x;
    const float* src = cb + (long)b * CODE_DIM;
    _Float16*    dst = cbh + (long)b * CODE_DIM;
    float s = 0.f;
    for (int c = threadIdx.x; c < CODE_DIM; c += 128) {
        float v = src[c];
        dst[c] = (_Float16)v;
        s += v * v;
    }
    __shared__ float red[128];
    red[threadIdx.x] = s; __syncthreads();
    for (int o = 64; o > 0; o >>= 1) {
        if (threadIdx.x < o) red[threadIdx.x] += red[threadIdx.x + o];
        __syncthreads();
    }
    if (threadIdx.x == 0) cnorm[b] = red[0];
}

__global__ void fill0_f(float* p, long n) {
    long i = (long)blockIdx.x * 256 + threadIdx.x;
    if (i < n) p[i] = 0.f;
}
__global__ void fill0_i(int* p, long n) {
    long i = (long)blockIdx.x * 256 + threadIdx.x;
    if (i < n) p[i] = 0;
}
__global__ void cvt_f2h(const float* __restrict__ s, _Float16* __restrict__ d, long n) {
    long i = (long)blockIdx.x * 256 + threadIdx.x;
    if (i < n) d[i] = (_Float16)s[i];
}

// ------------------------- fused conv1d GEMM (WMMA) -------------------------
//
// One wave computes a 64(Cout) x 16(B*Tout) tile: 4 M-subtiles (4 independent
// accumulator chains) sharing a single im2col B fragment each K-step.
// K = Kw * Cin_pad, stepped 32.
// A (weights): lane l (M row), half h: runs at
//   kb+16h and kb+16h+8. B (im2col input): lane l (N=n0+l), half h: runs at
//   kb+16h and kb+16h+8 (contiguous channels; C_in_pad%8==0 keeps runs inside
//   a single kernel tap).
// OUT_MODE: 0 = f16 channels-last, 1 = f32 (bounds-checked rows), 2 = both.
template<bool RELU_IN, bool RELU_OUT, bool RES_ADD, bool UPSAMPLE, int OUT_MODE>
__global__ __launch_bounds__(32) void conv_wmma_k(
    const _Float16* __restrict__ Xh,   // (B, Tin, Cin) f16 (Cin = padded)
    const _Float16* __restrict__ Wh,   // (Cout, Kw, Cin) f16
    const float*    __restrict__ bias, // Cout f32
    const _Float16* __restrict__ Resh, // residual, (B, Tout, Cout) f16
    _Float16*       __restrict__ Yh,   // f16 out
    float*          __restrict__ Yf,   // f32 out
    int Cin, int Tin, int Cout, int Tout, int Kw, int stride, int pad, int dil) {
    const int lane = threadIdx.x;
    const int l = lane & 15, h = lane >> 4;
    const int m0 = blockIdx.y * 64;
    const int n  = blockIdx.x * 16 + l;
    const int nb = n / Tout;
    const int t  = n - nb * Tout;
    const int Ktot = Kw * Cin;

    const _Float16* Arow[4];
    bool mValid[4];
#pragma unroll
    for (int mt = 0; mt < 4; mt++) {
        int mA = m0 + mt * 16 + l;
        mValid[mt] = (mA < Cout);
        Arow[mt]   = Wh + (long)mA * Ktot;
    }
    const _Float16* Xrow = Xh + (long)nb * Tin * Cin;

    v8f acc[4] = {};
    for (int kb = 0; kb < Ktot; kb += 32) {
        // ---- shared B fragment (im2col gather, once per 4 WMMAs) ----
        v16h b;
#pragma unroll
        for (int rr = 0; rr < 2; rr++) {
            int k = kb + 16 * h + 8 * rr;
            v8h bv = {};
            if (k < Ktot) {
                int kw = k / Cin;
                int ci = k - kw * Cin;
                int tin; bool ok;
                if (UPSAMPLE) {
                    int tu = t - pad + kw;          // on 2x-upsampled axis
                    ok  = (tu >= 0) && (tu < (Tin << 1));
                    tin = tu >> 1;
                } else {
                    tin = t * stride - pad + kw * dil;
                    ok  = (tin >= 0) && (tin < Tin);
                }
                if (ok) bv = *(const v8h*)(Xrow + (long)tin * Cin + ci);
            }
            if (RELU_IN) {
#pragma unroll
                for (int i = 0; i < 8; i++)
                    bv[i] = bv[i] > (_Float16)0.f ? bv[i] : (_Float16)0.f;
            }
#pragma unroll
            for (int i = 0; i < 8; i++) b[8 * rr + i] = bv[i];
        }
        // ---- 4 A fragments + 4 independent WMMAs ----
        const int ka0 = kb + 16 * h, ka1 = kb + 16 * h + 8;
        const bool k0ok = (ka0 < Ktot), k1ok = (ka1 < Ktot);
#pragma unroll
        for (int mt = 0; mt < 4; mt++) {
            v16h a;
            v8h a0 = {}, a1 = {};
            if (mValid[mt] && k0ok) a0 = *(const v8h*)(Arow[mt] + ka0);
            if (mValid[mt] && k1ok) a1 = *(const v8h*)(Arow[mt] + ka1);
#pragma unroll
            for (int i = 0; i < 8; i++) { a[i] = a0[i]; a[8 + i] = a1[i]; }
            acc[mt] = __builtin_amdgcn_wmma_f32_16x16x32_f16(
                false, a, false, b, (short)0, acc[mt], false, false);
        }
    }

    // ---- epilogue: bias, residual, relu, store (per M-subtile) ----
#pragma unroll
    for (int mt = 0; mt < 4; mt++) {
        const int mo = m0 + mt * 16 + 8 * h;   // rows mo..mo+7 (C/D layout)
        float vals[8];
#pragma unroll
        for (int r = 0; r < 8; r++) {
            int m = mo + r;
            vals[r] = (m < Cout) ? (acc[mt][r] + bias[m]) : 0.f;
        }
        if (RES_ADD) {
            v8h rv = *(const v8h*)(Resh + ((long)nb * Tout + t) * Cout + mo);
#pragma unroll
            for (int r = 0; r < 8; r++) vals[r] += (float)rv[r];
        }
        if (RELU_OUT) {
#pragma unroll
            for (int r = 0; r < 8; r++) vals[r] = fmaxf(vals[r], 0.f);
        }
        if (OUT_MODE == 0 || OUT_MODE == 2) {
            v8h o;
#pragma unroll
            for (int r = 0; r < 8; r++) o[r] = (_Float16)vals[r];
            *(v8h*)(Yh + ((long)nb * Tout + t) * Cout + mo) = o;
        }
        if (OUT_MODE == 1 || OUT_MODE == 2) {
            float* yp = Yf + ((long)nb * Tout + t) * Cout + mo;
#pragma unroll
            for (int r = 0; r < 8; r++)
                if (mo + r < Cout) yp[r] = vals[r];
        }
    }
}

// ----------------------------- VQ: argmin stage -----------------------------
// One wave per 16-token tile; sweeps the 64 code tiles two at a time (two
// independent WMMA chains sharing the resident A fragments).
__global__ __launch_bounds__(32) void vq_argmin(
    const _Float16* __restrict__ Rh,    // (2048, 512) residual f16
    const _Float16* __restrict__ CBh,   // (1024, 512) stage codebook f16
    const float*    __restrict__ cnorm, // 1024 code squared norms
    int*            __restrict__ idxOut) {
    const int lane = threadIdx.x;
    const int l = lane & 15, h = lane >> 4;
    const int m0 = blockIdx.x * 16;
    const _Float16* Arow = Rh + (long)(m0 + l) * CODE_DIM;

    float best[8]; int bidx[8];
#pragma unroll
    for (int r = 0; r < 8; r++) { best[r] = 3.402823466e38f; bidx[r] = 0; }

    for (int ct = 0; ct < NB_CODE / 16; ct += 2) {
        const int code0 = ct * 16 + l;
        const int code1 = code0 + 16;
        const _Float16* Brow0 = CBh + (long)code0 * CODE_DIM;
        const _Float16* Brow1 = CBh + (long)code1 * CODE_DIM;
        v8f acc0 = {}, acc1 = {};
#pragma unroll
        for (int kb = 0; kb < CODE_DIM; kb += 32) {
            v16h a, b0, b1;
            v8h av0 = *(const v8h*)(Arow + kb + 16 * h);
            v8h av1 = *(const v8h*)(Arow + kb + 16 * h + 8);
            v8h bv00 = *(const v8h*)(Brow0 + kb + 16 * h);
            v8h bv01 = *(const v8h*)(Brow0 + kb + 16 * h + 8);
            v8h bv10 = *(const v8h*)(Brow1 + kb + 16 * h);
            v8h bv11 = *(const v8h*)(Brow1 + kb + 16 * h + 8);
#pragma unroll
            for (int i = 0; i < 8; i++) {
                a[i] = av0[i];  a[8 + i] = av1[i];
                b0[i] = bv00[i]; b0[8 + i] = bv01[i];
                b1[i] = bv10[i]; b1[8 + i] = bv11[i];
            }
            acc0 = __builtin_amdgcn_wmma_f32_16x16x32_f16(
                false, a, false, b0, (short)0, acc0, false, false);
            acc1 = __builtin_amdgcn_wmma_f32_16x16x32_f16(
                false, a, false, b1, (short)0, acc1, false, false);
        }
        const float cn0 = cnorm[code0];
        const float cn1 = cnorm[code1];
#pragma unroll
        for (int r = 0; r < 8; r++) {
            float d0 = cn0 - 2.f * acc0[r];    // ||r||^2 constant per token
            if (d0 < best[r] || (d0 == best[r] && code0 < bidx[r])) {
                best[r] = d0; bidx[r] = code0;
            }
            float d1 = cn1 - 2.f * acc1[r];
            if (d1 < best[r] || (d1 == best[r] && code1 < bidx[r])) {
                best[r] = d1; bidx[r] = code1;
            }
        }
    }
    // reduce across the 16 lanes of each half (token = r + 8*h)
    for (int off = 1; off < 16; off <<= 1) {
#pragma unroll
        for (int r = 0; r < 8; r++) {
            float od = __shfl_xor(best[r], off, 32);
            int   oi = __shfl_xor(bidx[r], off, 32);
            if (od < best[r] || (od == best[r] && oi < bidx[r])) {
                best[r] = od; bidx[r] = oi;
            }
        }
    }
    if (l == 0) {
#pragma unroll
        for (int r = 0; r < 8; r++) idxOut[m0 + r + 8 * h] = bidx[r];
    }
}

// ------------------- VQ: residual/quantized update + stats ------------------
__global__ __launch_bounds__(256) void vq_update(
    const float* __restrict__ CB,      // (1024, 512) f32 stage codebook
    const int*   __restrict__ idx,     // 2048 indices for this stage
    float*       __restrict__ resid,   // (2048, 512) f32, updated in place
    _Float16*    __restrict__ residh,  // f16 mirror
    float*       __restrict__ quant,   // (2048, 512) f32 accumulator
    float*       __restrict__ commit_acc,
    int*         __restrict__ hist) {  // 1024 counts for this stage
    const int m = blockIdx.x;
    const int c = threadIdx.x;
    const int code = idx[m];
    float s = 0.f;
    for (int cc = c; cc < CODE_DIM; cc += 256) {
        float xq = CB[(long)code * CODE_DIM + cc];
        long  o  = (long)m * CODE_DIM + cc;
        float r  = resid[o];
        float d  = r - xq;
        s += d * d;
        resid[o]  = d;
        residh[o] = (_Float16)d;
        quant[o] += xq;
    }
    __shared__ float red[256];
    red[c] = s; __syncthreads();
    for (int o = 128; o > 0; o >>= 1) {
        if (c < o) red[c] += red[c + o];
        __syncthreads();
    }
    if (c == 0) {
        atomicAdd(commit_acc, red[0]);
        atomicAdd(&hist[code], 1);
    }
}

// -------------------------- VQ: loss / perplexity ---------------------------
__global__ __launch_bounds__(256) void vq_finalize(
    const int* __restrict__ hist, const float* __restrict__ commit_acc,
    float* __restrict__ out2) {   // out2[0]=commit_loss, out2[1]=perplexity
    __shared__ float red[256];
    __shared__ float perp_sum;
    if (threadIdx.x == 0) perp_sum = 0.f;
    __syncthreads();
    for (int q = 0; q < NUM_Q; q++) {
        float s = 0.f;
        for (int cc = threadIdx.x; cc < NB_CODE; cc += 256) {
            float p = (float)hist[q * NB_CODE + cc] * (1.f / (float)NTOK);
            s += p * logf(p + 1e-10f);
        }
        red[threadIdx.x] = s; __syncthreads();
        for (int o = 128; o > 0; o >>= 1) {
            if (threadIdx.x < o) red[threadIdx.x] += red[threadIdx.x + o];
            __syncthreads();
        }
        if (threadIdx.x == 0) perp_sum += expf(-red[0]);
        __syncthreads();
    }
    if (threadIdx.x == 0) {
        out2[0] = commit_acc[0] * (1.f / ((float)NTOK * (float)CODE_DIM));
        out2[1] = perp_sum * (1.f / (float)NUM_Q);
    }
}

// ------------------------------ host plumbing -------------------------------

struct ConvW {
    const float* bias;
    _Float16*    wp;     // packed (Cout, Kw, Cinp) f16
    int Cout, Cinp, Kw;
};

template<bool RI, bool RO, bool RA, bool UP, int OM>
static void run_conv(const _Float16* X, const ConvW& W, const _Float16* Res,
                     _Float16* Yh, float* Yf, int Tin, int Tout,
                     int stride, int pad, int dil, hipStream_t s) {
    dim3 grid((unsigned)(NBATCH * Tout / 16), (unsigned)((W.Cout + 63) / 64));
    conv_wmma_k<RI, RO, RA, UP, OM><<<grid, 32, 0, s>>>(
        X, W.wp, W.bias, Res, Yh, Yf, W.Cinp, Tin, W.Cout, Tout,
        W.Kw, stride, pad, dil);
}

extern "C" void kernel_launch(void* const* d_in, const int* in_sizes, int n_in,
                              void* d_out, int out_size, void* d_ws, size_t ws_size,
                              hipStream_t stream) {
    (void)in_sizes; (void)n_in; (void)out_size; (void)ws_size;
    auto F = [&](int i) { return (const float*)d_in[i]; };

    // Input order: setup_inputs dict order {x, params}; params leaves in
    // jax pytree order (dict keys sorted): codebooks, dec{c1_b,c1_w,c2_b,c2_w,
    // in_b,in_w,up[3]{res[3]{b1,b2,w1,w2},ub,uw}}, enc{down[3]{db,dw,
    // res[3]{b1,b2,w1,w2}},in_b,in_w,out_b,out_w}.
    const float* x_in      = F(0);
    const float* codebooks = F(1);

    // ---- workspace carve-up (256B aligned) ----
    size_t wsOff = 0;
    auto wsAlloc = [&](size_t bytes) -> void* {
        void* p = (char*)d_ws + wsOff;
        wsOff = (wsOff + bytes + 255) & ~(size_t)255;
        return p;
    };
    _Float16* xpack  = (_Float16*)wsAlloc((size_t)NBATCH * TFULL * INWP * 2);
    _Float16* actA   = (_Float16*)wsAlloc((size_t)NBATCH * TFULL * WIDTH * 2);
    _Float16* actB   = (_Float16*)wsAlloc((size_t)NBATCH * TFULL * WIDTH * 2);
    _Float16* actC   = (_Float16*)wsAlloc((size_t)NBATCH * TFULL * WIDTH * 2);
    float*    resid  = (float*)   wsAlloc((size_t)NTOK * CODE_DIM * 4);
    _Float16* residh = (_Float16*)wsAlloc((size_t)NTOK * CODE_DIM * 2);
    float*    quant  = (float*)   wsAlloc((size_t)NTOK * CODE_DIM * 4);
    _Float16* quanth = (_Float16*)wsAlloc((size_t)NTOK * CODE_DIM * 2);
    _Float16* cbh    = (_Float16*)wsAlloc((size_t)NUM_Q * NB_CODE * CODE_DIM * 2);
    float*    cnorm  = (float*)   wsAlloc((size_t)NUM_Q * NB_CODE * 4);
    int*      idxbuf = (int*)     wsAlloc((size_t)NUM_Q * NTOK * 4);
    int*      hist   = (int*)     wsAlloc((size_t)NUM_Q * NB_CODE * 4);
    float*    commit = (float*)   wsAlloc(64);

    // ---- weight packing (f32 -> padded f16, K-major per tap) ----
    auto prepW = [&](const float* w, const float* b, int Cout, int Cin, int Kw) -> ConvW {
        int  Cinp = (Cin + 7) & ~7;
        long tot  = (long)Cout * Kw * Cinp;
        _Float16* wp = (_Float16*)wsAlloc((size_t)tot * 2);
        prep_weight<<<(unsigned)((tot + 255) / 256), 256, 0, stream>>>(
            w, wp, Cout, Cin, Kw, Cinp, tot);
        return ConvW{b, wp, Cout, Cinp, Kw};
    };

    ConvW w_enc_in = prepW(F(93), F(92), WIDTH, INW, 3);
    ConvW w_enc_down[3], w_enc_r1[3][3], w_enc_r2[3][3];
    for (int b = 0; b < 3; b++) {
        int base = 50 + b * 14;
        w_enc_down[b] = prepW(F(base + 1), F(base + 0), WIDTH, WIDTH, 4);
        for (int j = 0; j < 3; j++) {
            int rb = base + 2 + j * 4;   // {b1,b2,w1,w2}
            w_enc_r1[b][j] = prepW(F(rb + 2), F(rb + 0), WIDTH, WIDTH, 3);
            w_enc_r2[b][j] = prepW(F(rb + 3), F(rb + 1), WIDTH, WIDTH, 1);
        }
    }
    ConvW w_enc_out = prepW(F(95), F(94), CODE_DIM, WIDTH, 3);
    ConvW w_dec_in  = prepW(F(7), F(6), WIDTH, CODE_DIM, 3);
    ConvW w_dec_r1[3][3], w_dec_r2[3][3], w_up[3];
    for (int b = 0; b < 3; b++) {
        int base = 8 + b * 14;
        for (int j = 0; j < 3; j++) {
            int rb = base + j * 4;
            w_dec_r1[b][j] = prepW(F(rb + 2), F(rb + 0), WIDTH, WIDTH, 3);
            w_dec_r2[b][j] = prepW(F(rb + 3), F(rb + 1), WIDTH, WIDTH, 1);
        }
        w_up[b] = prepW(F(base + 13), F(base + 12), WIDTH, WIDTH, 3);
    }
    ConvW w_c1 = prepW(F(3), F(2), WIDTH, WIDTH, 3);
    ConvW w_c2 = prepW(F(5), F(4), INW, WIDTH, 3);

    // ---- input pack + codebook prep + state init ----
    {
        long tot = (long)NBATCH * TFULL * INWP;
        pack_x<<<(unsigned)((tot + 255) / 256), 256, 0, stream>>>(x_in, xpack, tot);
        prep_cb<<<NUM_Q * NB_CODE, 128, 0, stream>>>(codebooks, cbh, cnorm);
        long nq = (long)NTOK * CODE_DIM;
        fill0_f<<<(unsigned)((nq + 255) / 256), 256, 0, stream>>>(quant, nq);
        fill0_i<<<(unsigned)((NUM_Q * NB_CODE + 255) / 256), 256, 0, stream>>>(
            hist, NUM_Q * NB_CODE);
        fill0_f<<<1, 256, 0, stream>>>(commit, 1);
    }

    // ---- encoder ----
    _Float16 *A = actA, *Bb = actB, *C = actC;
    run_conv<false, true, false, false, 0>(xpack, w_enc_in, nullptr, A, nullptr,
                                           TFULL, TFULL, 1, 1, 1, stream);
    int T = TFULL;
    for (int b = 0; b < 3; b++) {
        run_conv<false, false, false, false, 0>(A, w_enc_down[b], nullptr, Bb, nullptr,
                                                T, T / 2, 2, 1, 1, stream);
        T /= 2;
        { _Float16* t0 = A; A = Bb; Bb = t0; }   // h now in A
        int dil = 1;
        for (int j = 0; j < 3; j++) {
            run_conv<true, false, false, false, 0>(A, w_enc_r1[b][j], nullptr, C, nullptr,
                                                   T, T, 1, dil, dil, stream);
            run_conv<true, false, true, false, 0>(C, w_enc_r2[b][j], A, Bb, nullptr,
                                                  T, T, 1, 0, 1, stream);
            { _Float16* t0 = A; A = Bb; Bb = t0; }
            dil *= 3;
        }
    }
    // out conv -> tokens: f32 residual + f16 mirror (channels-last == (M, C))
    run_conv<false, false, false, false, 2>(A, w_enc_out, nullptr, residh, resid,
                                            TQ, TQ, 1, 1, 1, stream);

    // ---- residual VQ ----
    for (int q = 0; q < NUM_Q; q++) {
        vq_argmin<<<NTOK / 16, 32, 0, stream>>>(
            residh, cbh + (long)q * NB_CODE * CODE_DIM, cnorm + q * NB_CODE,
            idxbuf + q * NTOK);
        vq_update<<<NTOK, 256, 0, stream>>>(
            codebooks + (long)q * NB_CODE * CODE_DIM, idxbuf + q * NTOK,
            resid, residh, quant, commit, hist + q * NB_CODE);
    }
    {
        long nq = (long)NTOK * CODE_DIM;
        cvt_f2h<<<(unsigned)((nq + 255) / 256), 256, 0, stream>>>(quant, quanth, nq);
    }

    // ---- decoder ----
    run_conv<false, true, false, false, 0>(quanth, w_dec_in, nullptr, A, nullptr,
                                           TQ, TQ, 1, 1, 1, stream);
    T = TQ;
    for (int b = 0; b < 3; b++) {
        int dil = 9;
        for (int j = 0; j < 3; j++) {
            run_conv<true, false, false, false, 0>(A, w_dec_r1[b][j], nullptr, C, nullptr,
                                                   T, T, 1, dil, dil, stream);
            run_conv<true, false, true, false, 0>(C, w_dec_r2[b][j], A, Bb, nullptr,
                                                  T, T, 1, 0, 1, stream);
            { _Float16* t0 = A; A = Bb; Bb = t0; }
            dil /= 3;
        }
        // fused nearest-neighbor x2 upsample + conv
        run_conv<false, false, false, true, 0>(A, w_up[b], nullptr, Bb, nullptr,
                                               T, 2 * T, 1, 1, 1, stream);
        T *= 2;
        { _Float16* t0 = A; A = Bb; Bb = t0; }
    }
    run_conv<false, true, false, false, 0>(A, w_c1, nullptr, Bb, nullptr,
                                           TFULL, TFULL, 1, 1, 1, stream);
    // final conv writes f32 directly into d_out as (B, T, 263)
    float* out = (float*)d_out;
    run_conv<false, false, false, false, 1>(Bb, w_c2, nullptr, nullptr, out,
                                            TFULL, TFULL, 1, 1, 1, stream);

    // ---- losses -> last two output floats ----
    vq_finalize<<<1, 256, 0, stream>>>(hist, commit, out + XOUT_N);
}
